// IVPVAEODFlowPredictor_63737314672947
// MI455X (gfx1250) — compile-verified, matrix operand order checked by reading, not run on hardware
//
#include <hip/hip_runtime.h>

// ---------------------------------------------------------------------------
// IVP-VAE ODE flow predictor for gfx1250 (MI455X).
// B=512, T=8, D_IN=6, HID=128, LAT=64, OUT=2, DT=0.1
//
// ODE Euler steps use v_wmma_f32_16x16x32_f16.  Weight fragments AND the f32
// master z-state are register-resident; LDS holds only the f16 activation
// exchange buffers.  Each wave owns an N-column quadrant; 3 barriers/step.
// ---------------------------------------------------------------------------

#define B_   512
#define T_   8
#define DIN_ 6
#define HID_ 128
#define LAT_ 64
#define OUT_ 2

// padded LDS strides (f16 elements, multiples of 8 for 16B-aligned b128)
#define ZHP 72    // stride for z exchange buffer
#define HP  136   // stride for h1/h2

typedef __attribute__((ext_vector_type(16))) _Float16 v16h;
typedef __attribute__((ext_vector_type(8)))  _Float16 v8h;
typedef __attribute__((ext_vector_type(8)))  float    v8f;

// ---------------------------------------------------------------------------
__device__ __forceinline__ v8f wmma_f16(v16h a, v16h b, v8f c) {
  return __builtin_amdgcn_wmma_f32_16x16x32_f16(
      false, a, false, b, (short)0, c, false, false);
}

// Load a 16x32 f16 fragment (A-layout; B uses same layout on W^T).
// rowbase points at this lane's matrix row (row = tile*16 + (lane&15)).
__device__ __forceinline__ v16h frag_ld(const _Float16* rowbase, int k0, int half) {
  const _Float16* p = rowbase + k0 + half * 8;
  v8h lo = *(const v8h*)(p);
  v8h hi = *(const v8h*)(p + 16);
  v16h r;
#pragma unroll
  for (int i = 0; i < 8; ++i) { r[i] = lo[i]; r[i + 8] = hi[i]; }
  return r;
}

// ---------------------------------------------------------------------------
// One Euler step of dz/dt = MLP([z,t]) for a 64-row block.
//   h1 = tanh(z @ W1 + t*A1row64 + a1)   (K=64,  N=128)
//   h2 = tanh(h1 @ W2 + a2)              (K=128, N=128)
//   z += (h2 @ W3 + a3) * adt            (K=128, N=64)
// Wave q owns output columns [q*32,q*32+32) for L1/L2 and [q*16,q*16+16) for
// L3, iterating over all four 16-row M-tiles.  Weights, biases and the f32
// master z (zreg[m][v] <-> element (m*16+v+8*half, q*16+lrow)) live in
// registers; sZh is the f16 copy exchanged between waves.
// ---------------------------------------------------------------------------
__device__ __forceinline__ void ode_step(
    const v16h (&w1)[2][2], const v16h (&w2)[2][4], const v16h (&w3)[4],
    const float (&bb1)[2], const float (&rr)[2], const float (&bb2)[2],
    float bb3, float (&zreg)[4][8],
    _Float16* sZh, _Float16* sH1, _Float16* sH2,
    float tcur, float adt, int lane, int q) {
  const int half = lane >> 4;
  const int lrow = lane & 15;

  // ----- layer 1 -----
#pragma unroll
  for (int m = 0; m < 4; ++m) {
    const v16h a0 = frag_ld(sZh + (m * 16 + lrow) * ZHP, 0, half);
    const v16h a1 = frag_ld(sZh + (m * 16 + lrow) * ZHP, 32, half);
#pragma unroll
    for (int j = 0; j < 2; ++j) {
      v8f acc = {};
      acc = wmma_f16(a0, w1[j][0], acc);
      acc = wmma_f16(a1, w1[j][1], acc);
      const int col = (q * 2 + j) * 16 + lrow;
      const float bs = bb1[j] + tcur * rr[j];
#pragma unroll
      for (int v = 0; v < 8; ++v)
        sH1[(m * 16 + v + 8 * half) * HP + col] = (_Float16)tanhf(acc[v] + bs);
    }
  }
  __syncthreads();

  // ----- layer 2 -----
#pragma unroll
  for (int m = 0; m < 4; ++m) {
    v16h a[4];
#pragma unroll
    for (int kb = 0; kb < 4; ++kb)
      a[kb] = frag_ld(sH1 + (m * 16 + lrow) * HP, kb * 32, half);
#pragma unroll
    for (int j = 0; j < 2; ++j) {
      v8f acc = {};
#pragma unroll
      for (int kb = 0; kb < 4; ++kb) acc = wmma_f16(a[kb], w2[j][kb], acc);
      const int col = (q * 2 + j) * 16 + lrow;
#pragma unroll
      for (int v = 0; v < 8; ++v)
        sH2[(m * 16 + v + 8 * half) * HP + col] =
            (_Float16)tanhf(acc[v] + bb2[j]);
    }
  }
  __syncthreads();

  // ----- layer 3 + Euler update (register-resident f32 master state) -----
#pragma unroll
  for (int m = 0; m < 4; ++m) {
    v8f acc = {};
#pragma unroll
    for (int kb = 0; kb < 4; ++kb) {
      const v16h a = frag_ld(sH2 + (m * 16 + lrow) * HP, kb * 32, half);
      acc = wmma_f16(a, w3[kb], acc);
    }
    const int col = q * 16 + lrow;
#pragma unroll
    for (int v = 0; v < 8; ++v) {
      const float zn = zreg[m][v] + (acc[v] + bb3) * adt;
      zreg[m][v] = zn;
      sZh[(m * 16 + v + 8 * half) * ZHP + col] = (_Float16)zn;
    }
  }
  __syncthreads();
}

// Load this wave's persistent weight fragments + bias scalars from global.
__device__ __forceinline__ void load_wave_weights(
    const _Float16* gW1t, const _Float16* gW2t, const _Float16* gW3t,
    const float* gR64, const float* a1, const float* a2, const float* a3,
    int lane, int q,
    v16h (&w1)[2][2], v16h (&w2)[2][4], v16h (&w3)[4],
    float (&bb1)[2], float (&rr)[2], float (&bb2)[2], float& bb3) {
  const int half = lane >> 4;
  const int lrow = lane & 15;
#pragma unroll
  for (int j = 0; j < 2; ++j) {
    const int n = q * 2 + j;
#pragma unroll
    for (int kb = 0; kb < 2; ++kb)
      w1[j][kb] = frag_ld(gW1t + (n * 16 + lrow) * LAT_, kb * 32, half);
#pragma unroll
    for (int kb = 0; kb < 4; ++kb)
      w2[j][kb] = frag_ld(gW2t + (n * 16 + lrow) * HID_, kb * 32, half);
    const int col = n * 16 + lrow;
    bb1[j] = a1[col];
    rr[j]  = gR64[col];
    bb2[j] = a2[col];
  }
#pragma unroll
  for (int kb = 0; kb < 4; ++kb)
    w3[kb] = frag_ld(gW3t + (q * 16 + lrow) * HID_, kb * 32, half);
  bb3 = a3[q * 16 + lrow];
}

// ---------------------------------------------------------------------------
// Weight prep: transpose enc/dec A1/A2/A3 to f16 [N][K]; extract A1 row 64.
// ---------------------------------------------------------------------------
__global__ void prep_weights_kernel(
    const float* eA1, const float* eA2, const float* eA3,
    const float* dA1, const float* dA2, const float* dA3,
    _Float16* eW1t, _Float16* eW2t, _Float16* eW3t,
    _Float16* dW1t, _Float16* dW2t, _Float16* dW3t,
    float* eR64, float* dR64) {
  int i = blockIdx.x * blockDim.x + threadIdx.x;
  if (i < 8192) {                       // W1t [128][64] <- A1[65][128]
    int n = i >> 6, k = i & 63;
    eW1t[i] = (_Float16)eA1[k * 128 + n];
    dW1t[i] = (_Float16)dA1[k * 128 + n];
  } else if (i < 8192 + 16384) {        // W2t [128][128] <- A2[128][128]
    int j = i - 8192, n = j >> 7, k = j & 127;
    eW2t[j] = (_Float16)eA2[k * 128 + n];
    dW2t[j] = (_Float16)dA2[k * 128 + n];
  } else if (i < 8192 + 16384 + 8192) { // W3t [64][128] <- A3[128][64]
    int j = i - 8192 - 16384, n = j >> 7, k = j & 127;
    eW3t[j] = (_Float16)eA3[k * 64 + n];
    dW3t[j] = (_Float16)dA3[k * 64 + n];
  } else if (i < 32768 + 128) {         // t-column of A1
    int n = i - 32768;
    eR64[n] = eA1[64 * 128 + n];
    dR64[n] = dA1[64 * 128 + n];
  }
}

// ---------------------------------------------------------------------------
// Feature embedding: [B*T,6] -> 128 (lrelu) -> 64 (lrelu) -> 64.  1 row / WG.
// ---------------------------------------------------------------------------
__global__ __launch_bounds__(128) void embed_kernel(
    const float* feat, const float* W1, const float* b1,
    const float* W2, const float* b2, const float* W3, const float* b3,
    float* emb) {
  __shared__ float x[DIN_];
  __shared__ float h1[HID_];
  __shared__ float h2[LAT_];
  const int row = blockIdx.x, tid = threadIdx.x;
  if (tid < DIN_) x[tid] = feat[row * DIN_ + tid];
  __syncthreads();
  {
    float s = b1[tid];
    for (int k = 0; k < DIN_; ++k) s += x[k] * W1[k * HID_ + tid];
    h1[tid] = s >= 0.f ? s : 0.2f * s;
  }
  __syncthreads();
  if (tid < 64) {
    float s = b2[tid];
    for (int k = 0; k < HID_; ++k) s += h1[k] * W2[k * 64 + tid];
    h2[tid] = s >= 0.f ? s : 0.2f * s;
  }
  __syncthreads();
  if (tid < 64) {
    float s = b3[tid];
    for (int k = 0; k < 64; ++k) s += h2[k] * W3[k * 64 + tid];
    emb[row * LAT_ + tid] = s;
  }
}

// ---------------------------------------------------------------------------
// Encoder ODE: blockIdx = (tgroup-1)*8 + chunk.  tgroup (1..7) runs 10*t
// Euler steps with adt = -0.1 on 64 rows starting at chunk*64.
// ---------------------------------------------------------------------------
__global__ __launch_bounds__(128) void enc_ode_kernel(
    const float* emb,
    const _Float16* gW1t, const _Float16* gW2t, const _Float16* gW3t,
    const float* gR64, const float* a1, const float* a2, const float* a3,
    float* z0e) {
  __shared__ _Float16 sZh[64 * ZHP];
  __shared__ _Float16 sH1[64 * HP];
  __shared__ _Float16 sH2[64 * HP];

  const int tid = threadIdx.x, lane = tid & 31, q = tid >> 5;
  const int half = lane >> 4, lrow = lane & 15;
  v16h w1[2][2], w2[2][4], w3[4];
  float bb1[2], rr[2], bb2[2], bb3;
  load_wave_weights(gW1t, gW2t, gW3t, gR64, a1, a2, a3, lane, q,
                    w1, w2, w3, bb1, rr, bb2, bb3);

  const int tg = blockIdx.x / 8 + 1;      // time index 1..7
  const int b0 = (blockIdx.x % 8) * 64;   // first batch row

  // f16 exchange copy (cooperative) + f32 master state (per-lane registers)
  for (int i = tid; i < 64 * LAT_; i += 128) {
    int r = i >> 6, c = i & 63;
    sZh[r * ZHP + c] = (_Float16)emb[((b0 + r) * T_ + tg) * LAT_ + c];
  }
  float zreg[4][8];
  {
    const int c = q * 16 + lrow;
#pragma unroll
    for (int m = 0; m < 4; ++m)
#pragma unroll
      for (int v = 0; v < 8; ++v) {
        const int r = m * 16 + v + 8 * half;
        zreg[m][v] = emb[((b0 + r) * T_ + tg) * LAT_ + c];
      }
  }
  __syncthreads();

  const int steps = 10 * tg;
  for (int k = 0; k < steps; ++k)
    ode_step(w1, w2, w3, bb1, rr, bb2, bb3, zreg, sZh, sH1, sH2,
             0.1f * (float)k, -0.1f, lane, q);

  // write back from registers (each lane owns its elements)
  {
    const int c = q * 16 + lrow;
#pragma unroll
    for (int m = 0; m < 4; ++m)
#pragma unroll
      for (int v = 0; v < 8; ++v) {
        const int r = m * 16 + v + 8 * half;
        z0e[((b0 + r) * T_ + tg) * LAT_ + c] = zreg[m][v];
      }
  }
}

// ---------------------------------------------------------------------------
// Decoder ODE: one 70-step trajectory per 64 rows; snapshot every 10 steps.
// lat[:,0] = z0; lat[:,t] = state after 10*t steps (adt = +0.1).
// ---------------------------------------------------------------------------
__global__ __launch_bounds__(128) void dec_ode_kernel(
    const float* z0,
    const _Float16* gW1t, const _Float16* gW2t, const _Float16* gW3t,
    const float* gR64, const float* a1, const float* a2, const float* a3,
    float* lat) {
  __shared__ _Float16 sZh[64 * ZHP];
  __shared__ _Float16 sH1[64 * HP];
  __shared__ _Float16 sH2[64 * HP];

  const int tid = threadIdx.x, lane = tid & 31, q = tid >> 5;
  const int half = lane >> 4, lrow = lane & 15;
  v16h w1[2][2], w2[2][4], w3[4];
  float bb1[2], rr[2], bb2[2], bb3;
  load_wave_weights(gW1t, gW2t, gW3t, gR64, a1, a2, a3, lane, q,
                    w1, w2, w3, bb1, rr, bb2, bb3);

  const int b0 = blockIdx.x * 64;
  for (int i = tid; i < 64 * LAT_; i += 128) {
    int r = i >> 6, c = i & 63;
    float v = z0[(b0 + r) * LAT_ + c];
    sZh[r * ZHP + c] = (_Float16)v;
    lat[((b0 + r) * T_ + 0) * LAT_ + c] = v;     // t = 0 snapshot
  }
  float zreg[4][8];
  {
    const int c = q * 16 + lrow;
#pragma unroll
    for (int m = 0; m < 4; ++m)
#pragma unroll
      for (int v = 0; v < 8; ++v)
        zreg[m][v] = z0[(b0 + m * 16 + v + 8 * half) * LAT_ + c];
  }
  __syncthreads();

  for (int k = 0; k < 70; ++k) {
    ode_step(w1, w2, w3, bb1, rr, bb2, bb3, zreg, sZh, sH1, sH2,
             0.1f * (float)k, 0.1f, lane, q);
    if (((k + 1) % 10) == 0) {
      const int t = (k + 1) / 10;
      const int c = q * 16 + lrow;
#pragma unroll
      for (int m = 0; m < 4; ++m)
#pragma unroll
        for (int v = 0; v < 8; ++v) {
          const int r = m * 16 + v + 8 * half;
          lat[((b0 + r) * T_ + t) * LAT_ + c] = zreg[m][v];
        }
    }
  }
}

// ---------------------------------------------------------------------------
// Posterior net: z0e(row) -> lrelu(64) -> 128 => mu[row], lv[row].
// ---------------------------------------------------------------------------
__global__ __launch_bounds__(128) void posterior_kernel(
    const float* z0e, const float* emb,
    const float* pW1, const float* pb1, const float* pW2, const float* pb2,
    float* mu, float* lv) {
  __shared__ float zr[LAT_];
  __shared__ float q[LAT_];
  const int row = blockIdx.x, tid = threadIdx.x;
  const int t = row & (T_ - 1);
  const float* src = (t == 0) ? (emb + row * LAT_) : (z0e + row * LAT_);
  if (tid < LAT_) zr[tid] = src[tid];
  __syncthreads();
  if (tid < LAT_) {
    float s = pb1[tid];
    for (int k = 0; k < LAT_; ++k) s += zr[k] * pW1[k * LAT_ + tid];
    q[tid] = s >= 0.f ? s : 0.2f * s;
  }
  __syncthreads();
  float s = pb2[tid];
  for (int k = 0; k < LAT_; ++k) s += q[k] * pW2[k * 128 + tid];
  if (tid < LAT_) mu[row * LAT_ + tid] = s;
  else            lv[row * LAT_ + tid - LAT_] = s;
}

// ---------------------------------------------------------------------------
// Mixing weights (softmax over T) + reparameterization.  One thread per b.
// ---------------------------------------------------------------------------
__global__ void mix_kernel(
    const float* mu, const float* lv, const float* eps,
    const float* mW1, const float* mb1, const float* mW2, const float* mb2,
    float* z0) {
  const int b = blockIdx.x * blockDim.x + threadIdx.x;
  if (b >= B_) return;
  float wl[T_];
  for (int t = 0; t < T_; ++t) {
    const float* m = mu + (b * T_ + t) * LAT_;
    float s = mb2[0];
    for (int j = 0; j < 32; ++j) {
      float qv = mb1[j];
      for (int k = 0; k < LAT_; ++k) qv += m[k] * mW1[k * 32 + j];
      qv = qv >= 0.f ? qv : 0.2f * qv;
      s += qv * mW2[j];
    }
    wl[t] = s;
  }
  float mx = wl[0];
  for (int t = 1; t < T_; ++t) mx = fmaxf(mx, wl[t]);
  float den = 0.f;
  for (int t = 0; t < T_; ++t) { wl[t] = expf(wl[t] - mx); den += wl[t]; }
  const float inv = 1.f / den;
  for (int t = 0; t < T_; ++t) wl[t] *= inv;
  for (int c = 0; c < LAT_; ++c) {
    float m0 = 0.f, l0 = 0.f;
    for (int t = 0; t < T_; ++t) {
      m0 += wl[t] * mu[(b * T_ + t) * LAT_ + c];
      l0 += wl[t] * lv[(b * T_ + t) * LAT_ + c];
    }
    z0[b * LAT_ + c] = m0 + eps[b * LAT_ + c] * expf(0.5f * l0);
  }
}

// ---------------------------------------------------------------------------
// Reconstruction MLP: lat(row,64) -> 128 (lrelu) -> 64 (lrelu) -> 2.
// ---------------------------------------------------------------------------
__global__ __launch_bounds__(128) void recon_kernel(
    const float* lat, const float* W1, const float* b1,
    const float* W2, const float* b2, const float* W3, const float* b3,
    float* out) {
  __shared__ float x[LAT_];
  __shared__ float h1[HID_];
  __shared__ float h2[LAT_];
  const int row = blockIdx.x, tid = threadIdx.x;
  if (tid < LAT_) x[tid] = lat[row * LAT_ + tid];
  __syncthreads();
  {
    float s = b1[tid];
    for (int k = 0; k < LAT_; ++k) s += x[k] * W1[k * HID_ + tid];
    h1[tid] = s >= 0.f ? s : 0.2f * s;
  }
  __syncthreads();
  if (tid < LAT_) {
    float s = b2[tid];
    for (int k = 0; k < HID_; ++k) s += h1[k] * W2[k * LAT_ + tid];
    h2[tid] = s >= 0.f ? s : 0.2f * s;
  }
  __syncthreads();
  if (tid < OUT_) {
    float s = b3[tid];
    for (int k = 0; k < LAT_; ++k) s += h2[k] * W3[k * OUT_ + tid];
    out[row * OUT_ + tid] = s;
  }
}

// ---------------------------------------------------------------------------
// Host launch
// ---------------------------------------------------------------------------
extern "C" void kernel_launch(void* const* d_in, const int* in_sizes, int n_in,
                              void* d_out, int out_size, void* d_ws, size_t ws_size,
                              hipStream_t stream) {
  const float* features = (const float*)d_in[0];
  const float* eps      = (const float*)d_in[1];
  const float* fe_W1 = (const float*)d_in[2];  const float* fe_b1 = (const float*)d_in[3];
  const float* fe_W2 = (const float*)d_in[4];  const float* fe_b2 = (const float*)d_in[5];
  const float* fe_W3 = (const float*)d_in[6];  const float* fe_b3 = (const float*)d_in[7];
  const float* enc_A1 = (const float*)d_in[8];  const float* enc_a1 = (const float*)d_in[9];
  const float* enc_A2 = (const float*)d_in[10]; const float* enc_a2 = (const float*)d_in[11];
  const float* enc_A3 = (const float*)d_in[12]; const float* enc_a3 = (const float*)d_in[13];
  const float* post_W1 = (const float*)d_in[14]; const float* post_b1 = (const float*)d_in[15];
  const float* post_W2 = (const float*)d_in[16]; const float* post_b2 = (const float*)d_in[17];
  const float* mix_W1 = (const float*)d_in[18]; const float* mix_b1 = (const float*)d_in[19];
  const float* mix_W2 = (const float*)d_in[20]; const float* mix_b2 = (const float*)d_in[21];
  const float* dec_A1 = (const float*)d_in[22]; const float* dec_a1 = (const float*)d_in[23];
  const float* dec_A2 = (const float*)d_in[24]; const float* dec_a2 = (const float*)d_in[25];
  const float* dec_A3 = (const float*)d_in[26]; const float* dec_a3 = (const float*)d_in[27];
  const float* rec_W1 = (const float*)d_in[28]; const float* rec_b1 = (const float*)d_in[29];
  const float* rec_W2 = (const float*)d_in[30]; const float* rec_b2 = (const float*)d_in[31];
  const float* rec_W3 = (const float*)d_in[32]; const float* rec_b3 = (const float*)d_in[33];

  // Workspace layout (bytes)
  char* ws = (char*)d_ws;
  _Float16* eW1t = (_Float16*)(ws + 0);        // 16384 B
  _Float16* eW2t = (_Float16*)(ws + 16384);    // 32768 B
  _Float16* eW3t = (_Float16*)(ws + 49152);    // 16384 B
  _Float16* dW1t = (_Float16*)(ws + 65536);
  _Float16* dW2t = (_Float16*)(ws + 81920);
  _Float16* dW3t = (_Float16*)(ws + 114688);
  float* eR64 = (float*)(ws + 131072);
  float* dR64 = (float*)(ws + 131584);
  float* emb  = (float*)(ws + 132096);                      // 1 MiB
  float* z0e  = (float*)(ws + 132096 + 1048576);
  float* mu   = (float*)(ws + 132096 + 2 * 1048576);
  float* lv   = (float*)(ws + 132096 + 3 * 1048576);
  float* z0   = (float*)(ws + 132096 + 4 * 1048576);        // 128 KiB
  float* lat  = (float*)(ws + 132096 + 4 * 1048576 + 131072);
  float* out  = (float*)d_out;

  prep_weights_kernel<<<(32768 + 128 + 255) / 256, 256, 0, stream>>>(
      enc_A1, enc_A2, enc_A3, dec_A1, dec_A2, dec_A3,
      eW1t, eW2t, eW3t, dW1t, dW2t, dW3t, eR64, dR64);

  embed_kernel<<<B_ * T_, 128, 0, stream>>>(
      features, fe_W1, fe_b1, fe_W2, fe_b2, fe_W3, fe_b3, emb);

  enc_ode_kernel<<<7 * 8, 128, 0, stream>>>(
      emb, eW1t, eW2t, eW3t, eR64, enc_a1, enc_a2, enc_a3, z0e);

  posterior_kernel<<<B_ * T_, 128, 0, stream>>>(
      z0e, emb, post_W1, post_b1, post_W2, post_b2, mu, lv);

  mix_kernel<<<(B_ + 255) / 256, 256, 0, stream>>>(
      mu, lv, eps, mix_W1, mix_b1, mix_W2, mix_b2, z0);

  dec_ode_kernel<<<B_ / 64, 128, 0, stream>>>(
      z0, dW1t, dW2t, dW3t, dR64, dec_a1, dec_a2, dec_a3, lat);

  recon_kernel<<<B_ * T_, 128, 0, stream>>>(
      lat, rec_W1, rec_b1, rec_W2, rec_b2, rec_W3, rec_b3, out);
}
